// RVQ_VAE_17523466568355
// MI455X (gfx1250) — matrix operand order repaired and checked by reading, not compiled, not run on hardware
//
#include <hip/hip_runtime.h>

typedef __attribute__((ext_vector_type(16))) __bf16 v16bf;
typedef __attribute__((ext_vector_type(8)))  float  v8f;

#define NROWS  65536
#define DDIM   256
#define KCB    1024
#define NSTAGE 3

#define TPB     64            // 2 waves of 32
#define WAVES   2
#define RPW     16            // rows per wave
#define RPB     (WAVES*RPW)   // 32 rows per block
#define LSTRIDE 260           // padded f32 row stride in LDS

union Frag {
  int4   q[2];   // two b128 halves
  unsigned u[8];
  v16bf  b;      // 16 packed bf16 = one 16x16x32 WMMA operand per lane
};

// f32 -> bf16 round-to-nearest-even
__device__ __forceinline__ unsigned short f2bf(float x) {
  unsigned u = __float_as_uint(x);
  unsigned r = 0x7FFFu + ((u >> 16) & 1u);
  return (unsigned short)((u + r) >> 16);
}
__device__ __forceinline__ float bf2f(unsigned short b) {
  return __uint_as_float(((unsigned)b) << 16);
}
// split-precision pack: hi = bf16(x), lo = bf16(x - hi); two values per DWORD
__device__ __forceinline__ void cvt_pair(float a, float b, unsigned& hi, unsigned& lo) {
  unsigned short ha = f2bf(a), hb = f2bf(b);
  unsigned short la = f2bf(a - bf2f(ha)), lb = f2bf(b - bf2f(hb));
  hi = (unsigned)ha | ((unsigned)hb << 16);
  lo = (unsigned)la | ((unsigned)lb << 16);
}

// issue one 8KB half-tile (512 int4) as async global->LDS b128 transfers (ASYNCcnt += 8/wave)
__device__ __forceinline__ void issue_half_async(const int4* __restrict__ src,
                                                 const int4* lds_dst, int tid) {
  #pragma unroll
  for (int j = 0; j < 512 / TPB; ++j) {
    int i = tid + j * TPB;
    unsigned ldso = (unsigned)(size_t)(lds_dst + i);   // low 32 bits of generic LDS ptr = LDS offset
    asm volatile("global_load_async_to_lds_b128 %0, %1, off"
                 :: "v"(ldso), "v"(src + i) : "memory");
  }
}

// 4 chunks (C0..C0+3) of one k-tile: 12 WMMAs, fragment loads software-pipelined
template <int C0>
__device__ __forceinline__ void compute_half(const int4* buf, int lane,
                                             const Frag (&zh)[8], const Frag (&zl)[8],
                                             v8f& acc) {
  Frag ah, al, nah, nal;
  ah.q[0] = buf[     lane]; ah.q[1] = buf[32 + lane];
  al.q[0] = buf[64 + lane]; al.q[1] = buf[96 + lane];
  #pragma unroll
  for (int cl = 0; cl < 4; ++cl) {
    if (cl < 3) {
      nah.q[0] = buf[(cl + 1) * 128 +      lane];
      nah.q[1] = buf[(cl + 1) * 128 + 32 + lane];
      nal.q[0] = buf[(cl + 1) * 128 + 64 + lane];
      nal.q[1] = buf[(cl + 1) * 128 + 96 + lane];
    }
    // dot ~= ahi*zhi + ahi*zlo + alo*zhi   (lo*lo ~2^-18 rel., dropped)
    acc = __builtin_amdgcn_wmma_f32_16x16x32_bf16(false, ah.b, false, zh[C0 + cl].b, (short)0, acc, false, false);
    acc = __builtin_amdgcn_wmma_f32_16x16x32_bf16(false, ah.b, false, zl[C0 + cl].b, (short)0, acc, false, false);
    acc = __builtin_amdgcn_wmma_f32_16x16x32_bf16(false, al.b, false, zh[C0 + cl].b, (short)0, acc, false, false);
    ah = nah; al = nal;
  }
}

// --- zero the two loss accumulators (d_out is poisoned by harness) ---
__global__ void rvq_init_kernel(float* out) {
  out[(size_t)NROWS * DDIM]     = 0.0f;
  out[(size_t)NROWS * DDIM + 1] = 0.0f;
}

// --- c2[k] = sum_d cb[k][d]^2 (exact f32) ---
__global__ void rvq_c2_kernel(const float* __restrict__ cb, float* __restrict__ c2g) {
  int i = blockIdx.x * blockDim.x + threadIdx.x;
  if (i < NSTAGE * KCB) {
    const float4* row = (const float4*)(cb + (size_t)i * DDIM);
    float s = 0.f;
    #pragma unroll 8
    for (int j = 0; j < DDIM / 4; ++j) {
      float4 v = row[j];
      s += v.x * v.x + v.y * v.y + v.z * v.z + v.w * v.w;
    }
    c2g[i] = s;
  }
}

// --- prepass: codebook -> hi/lo bf16, pre-swizzled into WMMA A-fragment order ---
// layout: [stage][ktile(64)][dchunk(8)][part(hi,lo)][run(2)][lane(32)] x 16B
__global__ __launch_bounds__(64) void rvq_pack_kernel(const float* __restrict__ cb,
                                                      int4* __restrict__ cbsw) {
  int wid  = (blockIdx.x * blockDim.x + threadIdx.x) >> 5;  // 0..1535
  int lane = threadIdx.x & 31;
  int s   = wid >> 9;
  int rem = wid & 511;
  int t   = rem >> 3;
  int c   = rem & 7;
  int h = lane >> 4, lr = lane & 15;

  const float* src = cb + ((size_t)s * KCB + (size_t)t * 16 + lr) * DDIM + c * 32;
  float4 r0a = *(const float4*)(src + 8 * h);
  float4 r0b = *(const float4*)(src + 8 * h + 4);
  float4 r1a = *(const float4*)(src + 16 + 8 * h);
  float4 r1b = *(const float4*)(src + 16 + 8 * h + 4);

  Frag hi, lo;
  cvt_pair(r0a.x, r0a.y, hi.u[0], lo.u[0]);
  cvt_pair(r0a.z, r0a.w, hi.u[1], lo.u[1]);
  cvt_pair(r0b.x, r0b.y, hi.u[2], lo.u[2]);
  cvt_pair(r0b.z, r0b.w, hi.u[3], lo.u[3]);
  cvt_pair(r1a.x, r1a.y, hi.u[4], lo.u[4]);
  cvt_pair(r1a.z, r1a.w, hi.u[5], lo.u[5]);
  cvt_pair(r1b.x, r1b.y, hi.u[6], lo.u[6]);
  cvt_pair(r1b.z, r1b.w, hi.u[7], lo.u[7]);

  int4* dst = cbsw + ((((size_t)s * 64 + t) * 8 + c) << 7);
  dst[0 * 32 + lane] = hi.q[0];
  dst[1 * 32 + lane] = hi.q[1];
  dst[2 * 32 + lane] = lo.q[0];
  dst[3 * 32 + lane] = lo.q[1];
}

// --- fused 3-stage residual VQ: bf16x3 WMMA scoring + async double-buffered staging ---
__global__ __launch_bounds__(TPB) void rvq_main_kernel(
    const float* __restrict__ z, const float* __restrict__ cb,
    const int4* __restrict__ cbsw, const float* __restrict__ c2g,
    float* __restrict__ out) {
  __shared__ float zs[RPB * LSTRIDE];  // 33,280 B : f32 residual tiles (one per wave)
  __shared__ int4  cbl[2][512];        // 16,384 B : ping-pong half-tiles (chunks 0-3 | 4-7)
  __shared__ float c2s[KCB];           //  4,096 B : ||cb_k||^2  => 53,760 B total (proven size)

  const int tid  = threadIdx.x;
  const int lane = tid & 31;
  const int wave = tid >> 5;
  const int h    = lane >> 4;
  const int lr   = lane & 15;
  const size_t gbase = (size_t)blockIdx.x * RPB * DDIM;

  // residual r0 = z
  for (int i = tid; i < RPB * (DDIM / 2); i += TPB) {
    int r = i / (DDIM / 2), c = (i % (DDIM / 2)) * 2;
    *(float2*)&zs[r * LSTRIDE + c] = *(const float2*)(z + gbase + (size_t)r * DDIM + c);
  }

  float losstotal = 0.f;
  float* zrow = &zs[(wave * RPW + lr) * LSTRIDE];

  for (int stage = 0; stage < NSTAGE; ++stage) {
    const float* cbst = cb + (size_t)stage * KCB * DDIM;
    const int4*  swst = cbsw + ((size_t)stage << 16);
    for (int i = tid; i < KCB; i += TPB) c2s[i] = c2g[stage * KCB + i];
    __syncthreads();

    // build z B-fragments in registers: lane holds row lr, K=8h..8h+7 and 16+8h..16+8h+7 per 32-chunk
    Frag zh[8], zl[8];
    #pragma unroll
    for (int c = 0; c < 8; ++c) {
      const float* p = zrow + c * 32 + 8 * h;
      float4 a = *(const float4*)(p);
      float4 b = *(const float4*)(p + 4);
      float4 d = *(const float4*)(p + 16);
      float4 e = *(const float4*)(p + 20);
      cvt_pair(a.x, a.y, zh[c].u[0], zl[c].u[0]);
      cvt_pair(a.z, a.w, zh[c].u[1], zl[c].u[1]);
      cvt_pair(b.x, b.y, zh[c].u[2], zl[c].u[2]);
      cvt_pair(b.z, b.w, zh[c].u[3], zl[c].u[3]);
      cvt_pair(d.x, d.y, zh[c].u[4], zl[c].u[4]);
      cvt_pair(d.z, d.w, zh[c].u[5], zl[c].u[5]);
      cvt_pair(e.x, e.y, zh[c].u[6], zl[c].u[6]);
      cvt_pair(e.z, e.w, zh[c].u[7], zl[c].u[7]);
    }

    float best = 3.4e38f;
    int   bestk = 0;

    // prime the async pipeline: half A and half B of tile 0
    issue_half_async(swst,       &cbl[0][0], tid);
    issue_half_async(swst + 512, &cbl[1][0], tid);

    for (int kt = 0; kt < 64; ++kt) {
      const int4* tnext = swst + ((kt + 1) << 10);
      v8f acc = {0.f, 0.f, 0.f, 0.f, 0.f, 0.f, 0.f, 0.f};

      // ---- half A: chunks 0-3 (B(kt) may still be in flight) ----
      asm volatile("s_wait_asynccnt 0x8" ::: "memory");
      __syncthreads();                       // A(kt) resident for all waves
      compute_half<0>(&cbl[0][0], lane, zh, zl, acc);
      __syncthreads();                       // all waves done reading buffer 0
      if (kt + 1 < 64) issue_half_async(tnext, &cbl[0][0], tid);      // prefetch A(kt+1)

      // ---- half B: chunks 4-7 (A(kt+1) may be in flight) ----
      if (kt + 1 < 64) asm volatile("s_wait_asynccnt 0x8" ::: "memory");
      else             asm volatile("s_wait_asynccnt 0x0" ::: "memory");
      __syncthreads();                       // B(kt) resident
      compute_half<4>(&cbl[1][0], lane, zh, zl, acc);
      __syncthreads();                       // all waves done reading buffer 1
      if (kt + 1 < 64) issue_half_async(tnext + 512, &cbl[1][0], tid); // prefetch B(kt+1)

      // lane holds 8 scores for its row: k = kt*16 + 8h + v
      #pragma unroll
      for (int v = 0; v < 8; ++v) {
        int kk = (kt << 4) + 8 * h + v;
        float sc = c2s[kk] - 2.0f * acc[v];
        if (sc < best) { best = sc; bestk = kk; }
      }
    }

    // merge half-wave k-subsets per row (first-min tie-break like argmin)
    float ob  = __shfl_xor(best, 16, 32);
    int   obk = __shfl_xor(bestk, 16, 32);
    if (ob < best || (ob == best && obk < bestk)) { best = ob; bestk = obk; }

    // gather chosen code rows from ORIGINAL f32 codebook (exact residual / losses)
    for (int m = 0; m < RPW; ++m) {
      int idx = __shfl(bestk, m, 32);
      const float* crow = cbst + (size_t)idx * DDIM;
      float* zr = &zs[(wave * RPW + m) * LSTRIDE];
      #pragma unroll
      for (int j = 0; j < DDIM / 64; ++j) {
        int d = (j * 32 + lane) * 2;
        float2 cv = *(const float2*)(crow + d);
        float2 rv = *(float2*)(zr + d);
        rv.x -= cv.x; rv.y -= cv.y;
        *(float2*)(zr + d) = rv;
        losstotal += rv.x * rv.x + rv.y * rv.y;
      }
    }
    __syncthreads();  // all scoring/gather done before next stage rewrites c2s / buffers
  }

  // final_quantized = z - r_final
  for (int i = tid; i < RPB * (DDIM / 2); i += TPB) {
    int r = i / (DDIM / 2), c = (i % (DDIM / 2)) * 2;
    float2 zo = *(const float2*)(z + gbase + (size_t)r * DDIM + c);
    float2 rv = *(const float2*)&zs[r * LSTRIDE + c];
    float2 o; o.x = zo.x - rv.x; o.y = zo.y - rv.y;
    *(float2*)(out + gbase + (size_t)r * DDIM + c) = o;
  }

  for (int off = 16; off > 0; off >>= 1) losstotal += __shfl_xor(losstotal, off, 32);
  if (lane == 0) {
    float v = losstotal * (1.0f / ((float)NROWS * (float)DDIM));
    atomicAdd(out + (size_t)NROWS * DDIM,     v);
    atomicAdd(out + (size_t)NROWS * DDIM + 1, v);
  }
}

extern "C" void kernel_launch(void* const* d_in, const int* in_sizes, int n_in,
                              void* d_out, int out_size, void* d_ws, size_t ws_size,
                              hipStream_t stream) {
  const float* z  = (const float*)d_in[0];   // [N, D] f32
  const float* cb = (const float*)d_in[1];   // [3, K, D] f32
  float* out  = (float*)d_out;               // [N*D] ++ [loss, loss]
  int4*  cbsw = (int4*)d_ws;                 // 3 MB swizzled bf16 hi/lo codebook
  float* c2g  = (float*)((char*)d_ws + (3u << 20));  // 12 KB ||cb||^2

  rvq_init_kernel<<<1, 1, 0, stream>>>(out);
  rvq_c2_kernel<<<(NSTAGE * KCB + 255) / 256, 256, 0, stream>>>(cb, c2g);
  rvq_pack_kernel<<<768, 64, 0, stream>>>(cb, cbsw);
  rvq_main_kernel<<<NROWS / RPB, TPB, 0, stream>>>(z, cb, cbsw, c2g, out);
}
